// ElementLinear_17600775979356
// MI455X (gfx1250) — matrix operand order repaired and checked by reading, not compiled, not run on hardware
//
#include <hip/hip_runtime.h>
#include <hip/hip_bf16.h>

// ---------------------------------------------------------------------------
// ElementLinear: out[b,C,d] = sum_c (alpha * W[species[b],C,c]) * t[b,c,d]
// B=8192 nodes, Z=10 species, IN=OUT=512, M=3.
// Strategy: bin nodes by species -> per-species GEMM tiles on
// v_wmma_f32_16x16x32_bf16 (bf16 inputs, fp32 accumulate).
// ---------------------------------------------------------------------------

typedef __attribute__((ext_vector_type(16))) __bf16       v16bf;
typedef __attribute__((ext_vector_type(8)))  float        v8f;
typedef __attribute__((ext_vector_type(4)))  unsigned int v4u;

union FragU { v16bf bf; v4u u[2]; };

constexpr int Bn   = 8192;
constexpr int Zn   = 10;
constexpr int INC  = 512;
constexpr int OUTC = 512;
constexpr int Mm   = 3;
constexpr int GROUPS  = Bn / 16;          // 512 node-groups of 16 per species (capacity)
constexpr int NCOL    = 48;               // 16 nodes * 3 spherical components
constexpr int LSTRIDE = 520;              // LDS column stride (bf16 elems): bank-conflict-free
constexpr float ALPHA = 0.04419417382415922f;   // 1/sqrt(512)

__device__ __forceinline__ unsigned short f32_to_bf16_rne(float x) {
    unsigned u = __float_as_uint(x);
    u += 0x7FFFu + ((u >> 16) & 1u);      // round-to-nearest-even
    return (unsigned short)(u >> 16);
}

// ---------------- Phase 0: zero the per-species counters -------------------
__global__ void elin_zero(int* __restrict__ counts) {
    if (threadIdx.x < Zn) counts[threadIdx.x] = 0;
}

// ---------------- Phase 1: bin nodes by species ----------------------------
__global__ void elin_bin(const float* __restrict__ na,
                         int* __restrict__ counts, int* __restrict__ idx) {
    int b = blockIdx.x * 256 + threadIdx.x;
    if (b >= Bn) return;
    int z = 0;
    #pragma unroll
    for (int q = 0; q < Zn; ++q)
        if (na[b * Zn + q] > 0.5f) z = q;   // one-hot
    int pos = atomicAdd(&counts[z], 1);
    idx[z * Bn + pos] = b;
}

// ---- Phase 1b: convert weights (alpha folded) into WMMA A-fragment layout -
// Frag for (z, rowtile r, kstep s): 32 lanes x 16 bf16 contiguous (1KB).
// Lane L: M = L%16, koff = (L/16)*8; elem j -> K = s*32 + koff + (j<8 ? j : 8+j)
__global__ void elin_convw(const float* __restrict__ W,
                           unsigned short* __restrict__ Wbf) {
    int tid = blockIdx.x * 256 + threadIdx.x;
    if (tid >= Zn * OUTC * INC) return;
    int j = tid & 15;
    int L = (tid >> 4) & 31;
    int s = (tid >> 9) & 15;
    int r = (tid >> 13) & 31;
    int z = tid >> 18;
    int Mr   = L & 15;
    int koff = (L >> 4) * 8;
    int K    = s * 32 + koff + (j < 8 ? j : 8 + j);
    float v  = W[((size_t)z * OUTC + r * 16 + Mr) * INC + K] * ALPHA;
    Wbf[tid] = f32_to_bf16_rne(v);
}

// ---------------- Phase 2: per-group GEMM on WMMA --------------------------
__global__ __launch_bounds__(256) void elin_compute(
        const float* __restrict__ t,
        const int*   __restrict__ counts,
        const int*   __restrict__ idx,
        const unsigned short* __restrict__ Wbf,
        float* __restrict__ out) {
    const int z = blockIdx.x / GROUPS;
    const int g = blockIdx.x % GROUPS;
    const int cnt = counts[z];
    if (g * 16 >= cnt) return;                 // uniform across block
    const int nloc = min(16, cnt - g * 16);

    __shared__ __align__(16) unsigned short sT[NCOL * LSTRIDE]; // ~49.9 KB
    __shared__ int snodes[16];

    const int tid  = threadIdx.x;
    const int wave = tid >> 5;
    const int lane = tid & 31;

    if (tid < 16)
        snodes[tid] = (tid < nloc) ? idx[z * Bn + g * 16 + tid] : -1;
    __syncthreads();

    // ---- gather t[nodes] -> LDS bf16, column-major [col][K], col=(node,m) --
    // wave w loads nodes 2w and 2w+1; coalesced f32 global reads.
    #pragma unroll
    for (int jj = 0; jj < 2; ++jj) {
        const int j = wave * 2 + jj;
        const int b = snodes[j];
        #pragma unroll 4
        for (int i = 0; i < 48; ++i) {
            const int e = i * 32 + lane;               // 0..1535 = c*3+m
            const int c = e / 3;
            const int m = e - c * 3;
            float v = (b >= 0) ? t[(size_t)b * (INC * Mm) + e] : 0.0f;
            sT[(j * 3 + m) * LSTRIDE + c] = f32_to_bf16_rne(v);
        }
    }
    __syncthreads();

    // ---- WMMA main loop: this wave owns row tiles wave*4 .. wave*4+3 ------
    v8f acc[4][3];
    #pragma unroll
    for (int a = 0; a < 4; ++a)
        #pragma unroll
        for (int c = 0; c < 3; ++c) {
            #pragma unroll
            for (int q = 0; q < 8; ++q) acc[a][c][q] = 0.0f;
        }

    const int nIdx = lane & 15;
    const int kb   = (lane >> 4) * 8;

    for (int s = 0; s < 16; ++s) {             // K = 512 in steps of 32
        // B fragments from LDS (shared across the wave's 4 row tiles)
        FragU bf[3];
        #pragma unroll
        for (int ct = 0; ct < 3; ++ct) {
            const v4u* bp = (const v4u*)(&sT[(ct * 16 + nIdx) * LSTRIDE + s * 32 + kb]);
            bf[ct].u[0] = bp[0];
            bf[ct].u[1] = bp[2];
        }
        #pragma unroll
        for (int rt4 = 0; rt4 < 4; ++rt4) {
            const int rt = wave * 4 + rt4;
            FragU af;
            const v4u* ap = (const v4u*)(Wbf +
                ((size_t)((z * 32 + rt) * 16 + s) * 512 + lane * 16));
            af.u[0] = ap[0];
            af.u[1] = ap[1];
            #pragma unroll
            for (int ct = 0; ct < 3; ++ct) {
                acc[rt4][ct] = __builtin_amdgcn_wmma_f32_16x16x32_bf16(
                    /*neg_a=*/false, af.bf, /*neg_b=*/false, bf[ct].bf,
                    /*c_mod=*/(short)0, acc[rt4][ct],
                    /*reuse_a=*/false, /*reuse_b=*/false);
            }
        }
    }

    // ---- epilogue: D layout 16x16 f32 -> out[b, C, m] ----------------------
    const int mOff = (lane >> 4) * 8;          // M = vgpr + mOff
    const int N    = lane & 15;
    #pragma unroll
    for (int rt4 = 0; rt4 < 4; ++rt4) {
        const int rt = wave * 4 + rt4;
        #pragma unroll
        for (int ct = 0; ct < 3; ++ct) {
            const int col = ct * 16 + N;
            if (col < nloc * 3) {
                const int j = col / 3;
                const int m = col - j * 3;
                const int b = snodes[j];
                const size_t base = (size_t)b * (OUTC * Mm) + m;
                #pragma unroll
                for (int v = 0; v < 8; ++v) {
                    const int C = rt * 16 + mOff + v;
                    out[base + (size_t)C * 3] = acc[rt4][ct][v];
                }
            }
        }
    }
}

// ---------------------------------------------------------------------------
extern "C" void kernel_launch(void* const* d_in, const int* in_sizes, int n_in,
                              void* d_out, int out_size, void* d_ws, size_t ws_size,
                              hipStream_t stream) {
    const float* t  = (const float*)d_in[0];   // [B, IN, M] f32
    const float* na = (const float*)d_in[1];   // [B, Z] one-hot f32
    const float* W  = (const float*)d_in[2];   // [Z, OUT, IN] f32
    float* out = (float*)d_out;                // [B, OUT, M] f32

    char* ws = (char*)d_ws;
    int* counts = (int*)ws;                                 // Z ints
    int* idx    = (int*)(ws + 256);                         // B ints
    unsigned short* Wbf = (unsigned short*)(ws + 256 + sizeof(int) * Bn); // 5.24 MB bf16

    elin_zero<<<1, 32, 0, stream>>>(counts);
    elin_bin<<<(Bn + 255) / 256, 256, 0, stream>>>(na, counts, idx);
    elin_convw<<<(Zn * OUTC * INC + 255) / 256, 256, 0, stream>>>(W, Wbf);
    elin_compute<<<dim3(Zn * GROUPS), 256, 0, stream>>>(t, counts, idx, Wbf, out);
}
